// GraphSAGE_1872605741715
// MI455X (gfx1250) — compile-verified, matrix operand order checked by reading, not run on hardware
//
#include <hip/hip_runtime.h>
#include <hip/hip_bf16.h>
#include <stdint.h>

// ---- problem constants (match reference) ----
#define IN_FEATS  602
#define N_HIDDEN  256
#define N_CLASSES 41
#define N_SRC0    292864
#define N_DST0    11264
#define N_DST1    1024
#define FAN0      25
#define FAN1      10

#define K0PAD 608     // 602 padded to multiple of 32
#define K0TOT 1216    // [self(608) | neigh(608)] concatenated K
#define K1TOT 512     // [self(256) | neigh(256)]
#define NPAD1 48      // 41 padded to 3 tiles of 16

#define LROW 40       // LDS row stride in bf16 elems (80B: 16B-aligned, 20-bank stride)

typedef __attribute__((ext_vector_type(16))) __bf16 v16bf;
typedef __attribute__((ext_vector_type(8)))  float  v8f;
typedef __attribute__((__vector_size__(16))) int    vb128;   // builtin's pointee type

#ifndef __has_builtin
#define __has_builtin(x) 0
#endif
#if __has_builtin(__builtin_amdgcn_global_load_async_to_lds_b128)
#define HAVE_ASYNC_LDS 1
#else
#define HAVE_ASYNC_LDS 0
#endif

// ---- f32 <-> bf16 (round-to-nearest-even), bit-level ----
__device__ __forceinline__ unsigned short f2bf(float f) {
  union { float f; unsigned int u; } c; c.f = f;
  unsigned int u = c.u;
  return (unsigned short)((u + 0x7FFFu + ((u >> 16) & 1u)) >> 16);
}
__device__ __forceinline__ float bf2f(unsigned short h) {
  union { float f; unsigned int u; } c; c.u = ((unsigned int)h) << 16;
  return c.f;
}

// 16-byte global->LDS copy: async (ASYNCcnt-tracked, no VGPR staging) when
// available, else a plain vector load+store. AS(1)/AS(3) pointers built via
// integer casts: flat LDS address low 32 bits == LDS offset (ISA 10.2).
__device__ __forceinline__ void copy16(const unsigned short* gsrc, unsigned short* ldst) {
#if HAVE_ASYNC_LDS
  __builtin_amdgcn_global_load_async_to_lds_b128(
      (__attribute__((address_space(1))) vb128*)(uintptr_t)gsrc,
      (__attribute__((address_space(3))) vb128*)(unsigned int)(uintptr_t)ldst,
      0, 0);
#else
  *(uint4*)ldst = *(const uint4*)gsrc;
#endif
}
__device__ __forceinline__ void wait_async() {
#if HAVE_ASYNC_LDS
#if __has_builtin(__builtin_amdgcn_s_wait_asynccnt)
  __builtin_amdgcn_s_wait_asynccnt(0);
#else
  asm volatile("s_wait_asynccnt 0x0" ::: "memory");
#endif
#endif
}

// ---------------------------------------------------------------------------
// Pack [Wself0 ; pad ; Wneigh0 ; pad] transposed -> B0t[N_HIDDEN][K0TOT] bf16.
// ---------------------------------------------------------------------------
__global__ void pack_b0(const float* __restrict__ Wself, const float* __restrict__ Wneigh,
                        unsigned short* __restrict__ B0t) {
  int i = blockIdx.x * blockDim.x + threadIdx.x;
  if (i >= N_HIDDEN * K0TOT) return;
  int n = i / K0TOT;
  int k = i - n * K0TOT;
  float v = 0.0f;
  if (k < IN_FEATS)                              v = Wself[(long)k * N_HIDDEN + n];
  else if (k >= K0PAD && k < K0PAD + IN_FEATS)   v = Wneigh[(long)(k - K0PAD) * N_HIDDEN + n];
  B0t[i] = f2bf(v);
}

// ---------------------------------------------------------------------------
// Layer-0 aggregation: one block per dst node (coalesced 25-row gather-mean).
// ---------------------------------------------------------------------------
__global__ void agg0(const float* __restrict__ x, const int* __restrict__ idx,
                     unsigned short* __restrict__ A0) {
  const int d = blockIdx.x;
  const int t = threadIdx.x;            // 256 threads
  const long rowA = (long)d * K0TOT;
  const long rowX = (long)d * IN_FEATS;

  for (int c = t; c < IN_FEATS; c += 256)
    A0[rowA + c] = f2bf(x[rowX + c]);

  if (t < (K0PAD - IN_FEATS)) {         // 6 pad columns in each half
    A0[rowA + IN_FEATS + t] = 0;
    A0[rowA + K0PAD + IN_FEATS + t] = 0;
  }

  float s0 = 0.f, s1 = 0.f, s2 = 0.f;
  const int c0 = t, c1 = t + 256, c2 = t + 512;
  const int* nb = idx + d * FAN0;
  for (int j = 0; j < FAN0; ++j) {
    const float* xr = x + (long)nb[j] * IN_FEATS;
    s0 += xr[c0];
    if (c1 < IN_FEATS) s1 += xr[c1];
    if (c2 < IN_FEATS) s2 += xr[c2];
  }
  const float inv = 1.0f / (float)FAN0;
  A0[rowA + K0PAD + c0] = f2bf(s0 * inv);
  if (c1 < IN_FEATS) A0[rowA + K0PAD + c1] = f2bf(s1 * inv);
  if (c2 < IN_FEATS) A0[rowA + K0PAD + c2] = f2bf(s2 * inv);
}

// Fragment bit-container (two 16B halves = the two K-chunks of a lane).
union Frag { v16bf v; uint4 q[2]; };

// Load a 16x32 bf16 fragment from an LDS tile row-block (row stride LROW).
// ISA 7.12.2: lane = row (lanes 0-15) / same rows K+8 (lanes 16-31);
// per lane the 16 elems are K [8h..8h+7] and [16+8h..23+8h] -> two 16B reads.
__device__ __forceinline__ void lds_frag(const unsigned short* base, int row, int hv, Frag& f) {
  const unsigned short* p = base + row * LROW + 8 * hv;
  f.q[0] = *(const uint4*)(p);
  f.q[1] = *(const uint4*)(p + 16);
}

// ---------------------------------------------------------------------------
// GEMM0: H[11264,256] = relu(A0[11264,1216] @ B0 + b0), stored bf16.
// Block = 8 waves computes a 32(M) x 256(N) tile; per K-step the 32x32 A tile
// and 256x32 B tile are async-copied to LDS, then each wave runs a 2x2 grid
// of v_wmma_f32_16x16x32_bf16 (4 WMMA per A/B fragment pair).
// ---------------------------------------------------------------------------
__global__ void gemm0(const unsigned short* __restrict__ A0,
                      const unsigned short* __restrict__ B0t,
                      const float* __restrict__ b0,
                      unsigned short* __restrict__ H) {
  __shared__ unsigned short lsA[32 * LROW];       // 2.5 KB
  __shared__ unsigned short lsB[256 * LROW];      // 20 KB

  const int t    = threadIdx.x;                   // 256 threads = 8 waves
  const int wave = t >> 5;
  const int lane = t & 31;
  const int hv   = lane >> 4;
  const int r    = lane & 15;
  const int m0   = blockIdx.x * 32;

  v8f acc00 = {}, acc01 = {}, acc10 = {}, acc11 = {};

  for (int k0 = 0; k0 < K0TOT; k0 += 32) {
    // ---- stage B tile: 256 rows x 64B; thread t copies row n=t (4x16B) ----
    {
      const unsigned short* src = B0t + (size_t)t * K0TOT + k0;
      unsigned short* dst = lsB + t * LROW;
#pragma unroll
      for (int c = 0; c < 4; ++c) copy16(src + c * 8, dst + c * 8);
    }
    // ---- stage A tile: 32 rows x 64B; threads 0..127, one 16B chunk each ----
    if (t < 128) {
      const int row = t >> 2, c = t & 3;
      copy16(A0 + (size_t)(m0 + row) * K0TOT + k0 + c * 8, lsA + row * LROW + c * 8);
    }
    wait_async();            // each wave drains its own ASYNCcnt
    __syncthreads();         // tiles visible to all 8 waves

    Frag fa0, fa1, fb0, fb1;
    lds_frag(lsA, 0 + r, hv, fa0);                  // M-tile 0
    lds_frag(lsA, 16 + r, hv, fa1);                 // M-tile 1
    lds_frag(lsB, wave * 32 + r, hv, fb0);          // N-tile 2*wave
    lds_frag(lsB, wave * 32 + 16 + r, hv, fb1);     // N-tile 2*wave+1

    acc00 = __builtin_amdgcn_wmma_f32_16x16x32_bf16(false, fa0.v, false, fb0.v, (short)0, acc00, false, false);
    acc01 = __builtin_amdgcn_wmma_f32_16x16x32_bf16(false, fa0.v, false, fb1.v, (short)0, acc01, false, false);
    acc10 = __builtin_amdgcn_wmma_f32_16x16x32_bf16(false, fa1.v, false, fb0.v, (short)0, acc10, false, false);
    acc11 = __builtin_amdgcn_wmma_f32_16x16x32_bf16(false, fa1.v, false, fb1.v, (short)0, acc11, false, false);

    __syncthreads();         // protect LDS before next stage overwrites
  }

  // ---- epilogue: bias + relu + bf16 store. C/D layout: lane=N, VGPR r->M ----
  const int nA = wave * 32 + r;        // N col for fb0 tiles
  const int nB = nA + 16;              // N col for fb1 tiles
  const float biasA = b0[nA], biasB = b0[nB];
#pragma unroll
  for (int rr = 0; rr < 8; ++rr) {
    const int mA = m0 + rr + 8 * hv;       // M-tile 0 rows
    const int mB = mA + 16;                // M-tile 1 rows
    float v;
    v = acc00[rr] + biasA; v = v > 0.f ? v : 0.f; H[(long)mA * N_HIDDEN + nA] = f2bf(v);
    v = acc01[rr] + biasB; v = v > 0.f ? v : 0.f; H[(long)mA * N_HIDDEN + nB] = f2bf(v);
    v = acc10[rr] + biasA; v = v > 0.f ? v : 0.f; H[(long)mB * N_HIDDEN + nA] = f2bf(v);
    v = acc11[rr] + biasB; v = v > 0.f ? v : 0.f; H[(long)mB * N_HIDDEN + nB] = f2bf(v);
  }
}

// ---------------------------------------------------------------------------
// Layer-1 aggregation: A1[d] = [ H[d] | mean_10(H[nbr]) ] in bf16.
// ---------------------------------------------------------------------------
__global__ void agg1(const unsigned short* __restrict__ H, const int* __restrict__ idx,
                     unsigned short* __restrict__ A1) {
  const int d = blockIdx.x;
  const int c = threadIdx.x;             // 256 threads == N_HIDDEN
  A1[(long)d * K1TOT + c] = H[(long)d * N_HIDDEN + c];
  float s = 0.f;
  const int* nb = idx + d * FAN1;
  for (int j = 0; j < FAN1; ++j)
    s += bf2f(H[(long)nb[j] * N_HIDDEN + c]);
  A1[(long)d * K1TOT + N_HIDDEN + c] = f2bf(s * (1.0f / (float)FAN1));
}

// Pack [Wself1 ; Wneigh1] transposed -> B1t[NPAD1][K1TOT] bf16 (cols >= 41 zero).
__global__ void pack_b1(const float* __restrict__ Wself, const float* __restrict__ Wneigh,
                        unsigned short* __restrict__ B1t) {
  int i = blockIdx.x * blockDim.x + threadIdx.x;
  if (i >= NPAD1 * K1TOT) return;
  int n = i / K1TOT;
  int k = i - n * K1TOT;
  float v = 0.0f;
  if (n < N_CLASSES)
    v = (k < N_HIDDEN) ? Wself[(long)k * N_CLASSES + n]
                       : Wneigh[(long)(k - N_HIDDEN) * N_CLASSES + n];
  B1t[i] = f2bf(v);
}

// ---------------------------------------------------------------------------
// GEMM1 (tiny): one wave per 16x16 tile over K, fragments straight from
// global (L2-resident). A row-major, Bt = B transposed so cols contiguous.
// ---------------------------------------------------------------------------
__device__ __forceinline__ void wmma_tile(const unsigned short* __restrict__ A, int lda,
                                          const unsigned short* __restrict__ Bt, int ldb,
                                          int m0, int n0, int K, v8f& acc) {
  const int lane = threadIdx.x & 31;
  const int hv   = lane >> 4;
  const int rr   = lane & 15;
  const unsigned short* arow = A  + (long)(m0 + rr) * lda;
  const unsigned short* bcol = Bt + (long)(n0 + rr) * ldb;
  for (int k0 = 0; k0 < K; k0 += 32) {
    Frag a, b;
    a.q[0] = *(const uint4*)(arow + k0 + 8 * hv);
    a.q[1] = *(const uint4*)(arow + k0 + 16 + 8 * hv);
    b.q[0] = *(const uint4*)(bcol + k0 + 8 * hv);
    b.q[1] = *(const uint4*)(bcol + k0 + 16 + 8 * hv);
    acc = __builtin_amdgcn_wmma_f32_16x16x32_bf16(false, a.v, false, b.v, (short)0, acc, false, false);
  }
}

__global__ void gemm1(const unsigned short* __restrict__ A1,
                      const unsigned short* __restrict__ B1t,
                      const float* __restrict__ b1,
                      float* __restrict__ out) {
  const int w  = blockIdx.x * (blockDim.x >> 5) + (threadIdx.x >> 5);
  const int tn = w % 3;                  // 48/16 = 3 N tiles
  const int tm = w / 3;
  if (tm >= N_DST1 / 16) return;         // wave-uniform guard
  const int m0 = tm * 16, n0 = tn * 16;

  v8f acc = {};
  wmma_tile(A1, K1TOT, B1t, K1TOT, m0, n0, K1TOT, acc);

  const int lane = threadIdx.x & 31;
  const int nl = lane & 15, hv = lane >> 4;
  const int n = n0 + nl;
  if (n >= N_CLASSES) return;            // after WMMA: safe to diverge
  const float bias = b1[n];
#pragma unroll
  for (int rr = 0; rr < 8; ++rr) {
    const int m = m0 + rr + 8 * hv;
    out[(long)m * N_CLASSES + n] = acc[rr] + bias;
  }
}

// ---------------------------------------------------------------------------
extern "C" void kernel_launch(void* const* d_in, const int* in_sizes, int n_in,
                              void* d_out, int out_size, void* d_ws, size_t ws_size,
                              hipStream_t stream) {
  const float* x       = (const float*)d_in[0];
  const float* Wself0  = (const float*)d_in[1];
  const float* Wneigh0 = (const float*)d_in[2];
  const float* b0      = (const float*)d_in[3];
  const float* Wself1  = (const float*)d_in[4];
  const float* Wneigh1 = (const float*)d_in[5];
  const float* b1      = (const float*)d_in[6];
  const int*   idx0    = (const int*)d_in[7];
  const int*   idx1    = (const int*)d_in[8];
  float* out = (float*)d_out;

  // workspace layout (bf16 buffers, ~35 MB total)
  char* ws = (char*)d_ws;
  unsigned short* A0  = (unsigned short*)ws; ws += (size_t)N_DST0  * K0TOT    * 2;
  unsigned short* B0t = (unsigned short*)ws; ws += (size_t)N_HIDDEN * K0TOT   * 2;
  unsigned short* H   = (unsigned short*)ws; ws += (size_t)N_DST0  * N_HIDDEN * 2;
  unsigned short* A1  = (unsigned short*)ws; ws += (size_t)N_DST1  * K1TOT    * 2;
  unsigned short* B1t = (unsigned short*)ws;

  pack_b0<<<(N_HIDDEN * K0TOT + 255) / 256, 256, 0, stream>>>(Wself0, Wneigh0, B0t);
  agg0<<<N_DST0, 256, 0, stream>>>(x, idx0, A0);
  gemm0<<<N_DST0 / 32, 256, 0, stream>>>(A0, B0t, b0, H);
  agg1<<<N_DST1, 256, 0, stream>>>(H, idx1, A1);
  pack_b1<<<(NPAD1 * K1TOT + 255) / 256, 256, 0, stream>>>(Wself1, Wneigh1, B1t);
  gemm1<<<(N_DST1 / 16) * (NPAD1 / 16) / 8, 256, 0, stream>>>(A1, B1t, b1, out);
}
